// FasterRCNN_62165356642599
// MI455X (gfx1250) — matrix operand order repaired and checked by reading, not compiled, not run on hardware
//
#include <hip/hip_runtime.h>

// ---------- types ----------
typedef __bf16 v16bf __attribute__((ext_vector_type(16)));
typedef float  v8f   __attribute__((ext_vector_type(8)));
typedef unsigned int u32x4 __attribute__((ext_vector_type(4)));
typedef unsigned int v4u __attribute__((ext_vector_type(4)));
typedef int v4i __attribute__((ext_vector_type(4)));
typedef int v8i __attribute__((ext_vector_type(8)));

union BFrag { u32x4 q[2]; v16bf v; };

struct SPair { float s; int id; };

__device__ __forceinline__ unsigned short f2bf(float f) {
  unsigned u = __float_as_uint(f);
  unsigned r = (u + 0x7FFFu + ((u >> 16) & 1u)) >> 16;
  return (unsigned short)r;
}

// ---------- constants ----------
#define NPOS_TOTAL 40534      // 2 * (15200+3800+950+247+70)
#define LOG_MAX_F 4.135166556742356f
#define SORT_PER_IM 87296     // 65536+16384+4096+1024+256
#define NC 4096               // combined per-image slots (3951 used)

// workspace offsets (bytes)
#define OFF_WPACK  ((size_t)0)                         // 256*2304*2  = 1179648
#define OFF_PPACK  ((size_t)1179648)                   // 16*256*2    = 8192
#define OFF_H      ((size_t)(OFF_PPACK + 8192))        // 40534*256*2 = 20753408
#define OFF_PROJ   ((size_t)(OFF_H + 20753408))        // 40534*16*4  = 2594176
#define OFF_SORT   ((size_t)(OFF_PROJ + 2594176))      // 2*87296*8   = 1396736
#define OFF_CBOX   ((size_t)(OFF_SORT + 1396736))      // 2*4096*4*4  = 131072
#define OFF_CSCORE ((size_t)(OFF_CBOX + 131072))       // 32768
#define OFF_CGROUP ((size_t)(OFF_CSCORE + 32768))      // 32768
#define OFF_CVALID ((size_t)(OFF_CGROUP + 32768))      // 32768
#define OFF_NPAIR  ((size_t)(OFF_CVALID + 32768))      // 2*4096*8    = 65536
#define OFF_GB     ((size_t)(OFF_NPAIR + 65536))       // 2*4096*4*4  = 131072
#define OFF_GS     ((size_t)(OFF_GB + 131072))         // 32768
#define OFF_GG     ((size_t)(OFF_GS + 32768))          // 32768
#define OFF_GV     ((size_t)(OFF_GG + 32768))          // 32768

// ---------- TDM: async copy of a rows x 32(k) bf16 tile into padded LDS ------
// LDS rows: 64B data + 16B pad (=> 80B stride). pad_interval code 3 = 16 DWORDs,
// pad_amount code 3 = 4 DWORDs. data_size code 1 = 2 bytes.
__device__ __forceinline__ void tdm_load_A(unsigned lds_off, unsigned long long ga,
                                           int rows) {
  v4u g0;
  g0[0] = 1u;                                            // count=1, no gather
  g0[1] = lds_off;                                       // LDS byte address
  g0[2] = (unsigned)(ga & 0xFFFFFFFFull);                // global_addr[31:0]
  g0[3] = (unsigned)((ga >> 32) & 0x1FFFFFFull) | (2u << 30);  // addr[56:32] | type=2
  v8i g1;
  g1[0] = (int)((1u << 16) | (1u << 20) | (3u << 22) | (3u << 25)); // dsz=2B, pad en/int/amt
  g1[1] = (int)(2304u << 16);     // tensor_dim0[15:0] << 16 (dim0 = 2304)
  g1[2] = (int)(256u << 16);      // dim0[31:16]=0 | tensor_dim1[15:0]=256
  g1[3] = (int)(32u << 16);       // dim1[31:16]=0 | tile_dim0=32
  g1[4] = rows;                   // tile_dim1=rows | tile_dim2=0
  g1[5] = 2304;                   // tensor_dim0_stride[31:0]
  g1[6] = 0;                      // stride0[47:32] | stride1[15:0]
  g1[7] = 0;                      // stride1[47:16]
  v4i gz = {0, 0, 0, 0};
#if defined(__clang_major__) && (__clang_major__ >= 23)
  v8i gz8 = {0, 0, 0, 0, 0, 0, 0, 0};
  __builtin_amdgcn_tensor_load_to_lds(g0, g1, gz, gz, gz8, 0);
#else
  __builtin_amdgcn_tensor_load_to_lds(g0, g1, gz, gz, 0);
#endif
}

// ---------- weight packing ----------
__global__ __launch_bounds__(256) void pack_conv_w(const float* __restrict__ cw,
                                                   unsigned short* __restrict__ wpack) {
  int idx = blockIdx.x * 256 + threadIdx.x;          // 2304 blocks -> 589824
  int oc = idx / 2304;
  int k  = idx - oc * 2304;
  int tap = k >> 8;
  int ic  = k & 255;
  wpack[idx] = f2bf(cw[((size_t)oc * 256 + ic) * 9 + tap]);
}

__global__ __launch_bounds__(256) void pack_proj_w(const float* __restrict__ reg_w,
                                                   const float* __restrict__ log_w,
                                                   unsigned short* __restrict__ ppack) {
  int m = blockIdx.x;          // 16
  int ic = threadIdx.x;        // 256
  float v = 0.f;
  if (m < 12) v = reg_w[m * 256 + ic];
  else if (m < 15) v = log_w[(m - 12) * 256 + ic];
  ppack[m * 256 + ic] = f2bf(v);
}

// ---------- conv 3x3 as implicit GEMM, bf16 WMMA + TDM weight staging ----------
// M = 256 (oc), N = 2*H*W (pos), K = 2304.
// Block: 128oc x 64pos, 8 waves (4x2); each wave computes a 32oc x 32pos tile
// (4 WMMAs per K-chunk from 8 ds_load_b128).
__global__ __launch_bounds__(256)
void conv3x3_bf16_wmma(const float* __restrict__ fmap,
                       const unsigned short* __restrict__ wpack,
                       const float* __restrict__ conv_b,
                       unsigned short* __restrict__ hout,
                       int H, int W, int posCount, int posOff) {
  __shared__ unsigned short Ald[128 * 40];  // 128 oc x 32 k (+8 bf16 pad per row)
  __shared__ unsigned short Bld[64 * 40];   // 64 pos x 32 k (+8 bf16 pad per row)

  const int t = threadIdx.x;
  const int lane = t & 31, wave = t >> 5;
  const int waveOC = wave & 3, wavePos = wave >> 2;   // 4 x 2 wave grid
  const int posBase = blockIdx.x * 64;
  const int ocBase  = blockIdx.y * 128;
  const int HW = H * W;

  v8f acc00 = {0.f,0.f,0.f,0.f,0.f,0.f,0.f,0.f};
  v8f acc01 = {0.f,0.f,0.f,0.f,0.f,0.f,0.f,0.f};
  v8f acc10 = {0.f,0.f,0.f,0.f,0.f,0.f,0.f,0.f};
  v8f acc11 = {0.f,0.f,0.f,0.f,0.f,0.f,0.f,0.f};

  // B staging mapping: thread -> (pos_local, k-row)
  const int posl = t & 63;
  const int krow = t >> 6;
  const int p = posBase + posl;
  const bool pin = (p < posCount);
  int n = 0, y0 = 0, x0 = 0;
  if (pin) { n = p / HW; int rem = p - n * HW; y0 = rem / W; x0 = rem - y0 * W; }

  const unsigned ldsA = (unsigned)(size_t)(&Ald[0]);  // low 32 bits = LDS offset
  const unsigned long long wbase =
      (unsigned long long)(size_t)(wpack + (size_t)ocBase * 2304);

  const int mrow = lane & 15;
  const int khalf = lane >> 4;
  const unsigned short* ar0 = &Ald[(waveOC * 32 + mrow) * 40 + khalf * 8];
  const unsigned short* ar1 = ar0 + 16 * 40;
  const unsigned short* br0 = &Bld[(wavePos * 32 + mrow) * 40 + khalf * 8];
  const unsigned short* br1 = br0 + 16 * 40;

  for (int tap = 0; tap < 9; ++tap) {
    const int t3 = tap / 3;
    const int ky = t3 - 1;
    const int kx = (tap - t3 * 3) - 1;
    const int yy = y0 + ky, xx = x0 + kx;
    const bool inb = pin && ((unsigned)yy < (unsigned)H) && ((unsigned)xx < (unsigned)W);
    // Safe base: out-of-bounds lanes read inside fmap (result masked to 0).
    const float* pS = inb
        ? (fmap + (((size_t)n * 256) * H + yy) * W + xx + (size_t)krow * HW)
        : (fmap + (size_t)krow * HW);
    const size_t step = inb ? (size_t)32 * HW : 0;   // advance only real lanes
    const float mask = inb ? 1.f : 0.f;
    (void)mask;

    for (int icb = 0; icb < 256; icb += 32) {
      const int k0 = tap * 256 + icb;
      // stage A tile via Tensor Data Mover (wave 0 issues for the block)
      if (wave == 0) {
        tdm_load_A(ldsA, wbase + (unsigned long long)(k0) * 2u, 128);
      }
      if (icb < 224)
        __builtin_prefetch(pS + (size_t)32 * HW, 0, 1);   // next ic-chunk
      // stage B tile: unconditional strided loads, select-zero after
#pragma unroll
      for (int i = 0; i < 8; ++i) {
        float v = pS[(size_t)i * 4 * HW];
        v = inb ? v : 0.f;
        Bld[posl * 40 + ((i << 2) + krow)] = f2bf(v);
      }
      pS += step;
      if (wave == 0) __builtin_amdgcn_s_wait_tensorcnt(0);
      __syncthreads();

      // fragment loads per documented 16-bit A/B VGPR layouts
      BFrag a0, a1, b0, b1;
      a0.q[0] = *(const u32x4*)(ar0);           // K = khalf*8 .. +7
      a0.q[1] = *(const u32x4*)(ar0 + 16);      // K = 16 + khalf*8 .. +7
      a1.q[0] = *(const u32x4*)(ar1);
      a1.q[1] = *(const u32x4*)(ar1 + 16);
      b0.q[0] = *(const u32x4*)(br0);           // K = khalf*8 .. +7
      b0.q[1] = *(const u32x4*)(br0 + 16);      // K = 16 + khalf*8 .. +7
      b1.q[0] = *(const u32x4*)(br1);
      b1.q[1] = *(const u32x4*)(br1 + 16);

      acc00 = __builtin_amdgcn_wmma_f32_16x16x32_bf16(false, a0.v, false, b0.v,
                                                      (short)0, acc00, false, false);
      acc01 = __builtin_amdgcn_wmma_f32_16x16x32_bf16(false, a0.v, false, b1.v,
                                                      (short)0, acc01, false, false);
      acc10 = __builtin_amdgcn_wmma_f32_16x16x32_bf16(false, a1.v, false, b0.v,
                                                      (short)0, acc10, false, false);
      acc11 = __builtin_amdgcn_wmma_f32_16x16x32_bf16(false, a1.v, false, b1.v,
                                                      (short)0, acc11, false, false);
      __syncthreads();
    }
  }

  // epilogue: bias + relu -> bf16 h[pos][oc]
  const int ncol = lane & 15;
  const int mhi  = lane >> 4;
  const int posT0 = posBase + wavePos * 32 + ncol;
  const int posT1 = posT0 + 16;
  const int ocT0  = ocBase + waveOC * 32 + mhi * 8;
  const int ocT1  = ocT0 + 16;
#pragma unroll
  for (int r = 0; r < 8; ++r) {
    const float bias0 = conv_b[ocT0 + r];
    const float bias1 = conv_b[ocT1 + r];
    if (posT0 < posCount) {
      hout[(size_t)(posOff + posT0) * 256 + ocT0 + r] = f2bf(fmaxf(acc00[r] + bias0, 0.f));
      hout[(size_t)(posOff + posT0) * 256 + ocT1 + r] = f2bf(fmaxf(acc10[r] + bias1, 0.f));
    }
    if (posT1 < posCount) {
      hout[(size_t)(posOff + posT1) * 256 + ocT0 + r] = f2bf(fmaxf(acc01[r] + bias0, 0.f));
      hout[(size_t)(posOff + posT1) * 256 + ocT1 + r] = f2bf(fmaxf(acc11[r] + bias1, 0.f));
    }
  }
}

// ---------- projection: out[16][pos] = W(16x256) * h(256xpos), WMMA ----------
__global__ __launch_bounds__(256)
void proj_wmma(const unsigned short* __restrict__ h,
               const unsigned short* __restrict__ ppack,
               const float* __restrict__ reg_b,
               const float* __restrict__ log_b,
               float* __restrict__ proj, int P) {
  const int lane = threadIdx.x & 31, wave = threadIdx.x >> 5;
  const int posBase = (blockIdx.x * 8 + wave) * 16;
  const int mrow = lane & 15, khalf = lane >> 4;
  int pr = posBase + mrow; if (pr >= P) pr = P - 1;
  const unsigned short* hrow = h + (size_t)pr * 256;
  const unsigned short* arow = ppack + mrow * 256;
  v8f acc = {0.f,0.f,0.f,0.f,0.f,0.f,0.f,0.f};
#pragma unroll
  for (int k0 = 0; k0 < 256; k0 += 32) {
    BFrag a, b;
    a.q[0] = *(const u32x4*)(arow + k0 + khalf * 8);
    a.q[1] = *(const u32x4*)(arow + k0 + 16 + khalf * 8);
    b.q[0] = *(const u32x4*)(hrow + k0 + khalf * 8);
    b.q[1] = *(const u32x4*)(hrow + k0 + 16 + khalf * 8);
    acc = __builtin_amdgcn_wmma_f32_16x16x32_bf16(false, a.v, false, b.v,
                                                  (short)0, acc, false, false);
  }
  const int pos = posBase + (lane & 15);
  if (pos < P) {
#pragma unroll
    for (int r = 0; r < 8; ++r) {
      int m = khalf * 8 + r;
      float bias = (m < 12) ? reg_b[m] : ((m < 15) ? log_b[m - 12] : 0.f);
      proj[(size_t)pos * 16 + m] = acc[r] + bias;
    }
  }
}

// ---------- fill per-(img,level) sort buffers with logits ----------
__global__ __launch_bounds__(256)
void fill_sort(const float* __restrict__ proj, SPair* __restrict__ sortbuf) {
  int e = blockIdx.x * 256 + threadIdx.x;         // 174592 total
  int im = e / SORT_PER_IM;
  int rem = e - im * SORT_PER_IM;
  const int padOff[6] = {0, 65536, 81920, 86016, 87040, 87296};
  const int Anch[5]   = {45600, 11400, 2850, 741, 210};
  const int HWs[5]    = {15200, 3800, 950, 247, 70};
  const int posOff[5] = {0, 30400, 38000, 39900, 40394};
  int l = 4;
#pragma unroll
  for (int q = 0; q < 5; ++q)
    if (rem >= padOff[q] && rem < padOff[q + 1]) l = q;
  int slot = rem - padOff[l];
  SPair pr;
  if (slot < Anch[l]) {
    int cell = slot / 3;
    int r = slot - cell * 3;
    int g = posOff[l] + im * HWs[l] + cell;
    pr.s = proj[(size_t)g * 16 + 12 + r];
    pr.id = slot;
  } else {
    pr.s = -1e30f;
    pr.id = -1;
  }
  sortbuf[e] = pr;
}

// ---------- bitonic sort (descending) per (img,level) group ----------
__global__ __launch_bounds__(1024)
void sort_groups(SPair* __restrict__ buf) {
  const int padOff[5]  = {0, 65536, 81920, 86016, 87040};
  const int padSize[5] = {65536, 16384, 4096, 1024, 256};
  int gi = blockIdx.x;
  int im = gi / 5, l = gi - im * 5;
  SPair* a = buf + (size_t)im * SORT_PER_IM + padOff[l];
  unsigned Np = (unsigned)padSize[l];
  for (unsigned ksz = 2; ksz <= Np; ksz <<= 1) {
    for (unsigned j = ksz >> 1; j > 0; j >>= 1) {
      for (unsigned i = threadIdx.x; i < Np; i += blockDim.x) {
        unsigned q = i ^ j;
        if (q > i) {
          SPair A = a[i], B = a[q];
          bool desc = ((i & ksz) == 0);
          bool sw = desc ? (A.s < B.s) : (A.s > B.s);
          if (sw) { a[i] = B; a[q] = A; }
        }
      }
      __syncthreads();
    }
  }
}

// ---------- decode top-k anchors into combined per-image arrays ----------
__global__ __launch_bounds__(256)
void decode_combined(const SPair* __restrict__ sortbuf,
                     const float* __restrict__ proj,
                     const float* __restrict__ pri0, const float* __restrict__ pri1,
                     const float* __restrict__ pri2, const float* __restrict__ pri3,
                     const float* __restrict__ pri4,
                     const float* __restrict__ imsizes,
                     float* __restrict__ cbox, float* __restrict__ cscore,
                     int* __restrict__ cgroup, int* __restrict__ cvalid,
                     SPair* __restrict__ npair) {
  int e = blockIdx.x * 256 + threadIdx.x;   // 8192
  int im = e >> 12;
  int c = e & (NC - 1);
  const int lvbase[6] = {0, 1000, 2000, 3000, 3741, 3951};
  const int padOff[5] = {0, 65536, 81920, 86016, 87040};
  const int HWs[5]    = {15200, 3800, 950, 247, 70};
  const int posOff[5] = {0, 30400, 38000, 39900, 40394};
  const float* pris[5] = {pri0, pri1, pri2, pri3, pri4};
  int l = -1;
#pragma unroll
  for (int q = 0; q < 5; ++q)
    if (c >= lvbase[q] && c < lvbase[q + 1]) l = q;
  float sc = -1e30f, sortkey = -1e30f;
  float bx0 = 0.f, bx1 = 0.f, bx2 = 0.f, bx3 = 0.f;
  int grp = 0, val = 0;
  if (l >= 0) {
    int rank = c - lvbase[l];
    SPair p = sortbuf[(size_t)im * SORT_PER_IM + padOff[l] + rank];
    int s = p.id;
    float logit = p.s;
    int cell = s / 3;
    int r = s - cell * 3;
    int g = posOff[l] + im * HWs[l] + cell;
    const float* pri = pris[l] + (size_t)s * 4;
    float pcx = pri[0], pcy = pri[1], pw = pri[2], ph = pri[3];
    const float* rp = proj + (size_t)g * 16 + r * 4;
    float dx = rp[0], dy = rp[1];
    float dw = fminf(rp[2], LOG_MAX_F);
    float dh = fminf(rp[3], LOG_MAX_F);
    float cx = dx * pw + pcx, cy = dy * ph + pcy;
    float w = pw * __expf(dw), h = ph * __expf(dh);
    float him = imsizes[im * 2 + 0], wim = imsizes[im * 2 + 1];
    bx0 = fminf(fmaxf(cx - 0.5f * w, 0.f), wim);
    bx1 = fminf(fmaxf(cy - 0.5f * h, 0.f), him);
    bx2 = fminf(fmaxf(cx + 0.5f * w, 0.f), wim);
    bx3 = fminf(fmaxf(cy + 0.5f * h, 0.f), him);
    sc = 1.f / (1.f + __expf(-logit));
    val = ((bx2 - bx0) >= 0.001f) && ((bx3 - bx1) >= 0.001f);
    grp = im * 10 + l;
    sortkey = val ? sc : -1e30f;   // invalid -> sort last (reference: -inf)
  }
  int o = im * NC + c;
  cbox[o * 4 + 0] = bx0; cbox[o * 4 + 1] = bx1;
  cbox[o * 4 + 2] = bx2; cbox[o * 4 + 3] = bx3;
  cscore[o] = sc; cgroup[o] = grp; cvalid[o] = val;
  SPair np; np.s = sortkey; np.id = c;
  npair[o] = np;
}

// ---------- per-image bitonic sort of 4096 candidates (descending) ----------
__global__ __launch_bounds__(1024)
void sort_image(SPair* __restrict__ npair) {
  SPair* a = npair + (size_t)blockIdx.x * NC;
  for (unsigned ksz = 2; ksz <= NC; ksz <<= 1) {
    for (unsigned j = ksz >> 1; j > 0; j >>= 1) {
      for (unsigned i = threadIdx.x; i < NC; i += blockDim.x) {
        unsigned q = i ^ j;
        if (q > i) {
          SPair A = a[i], B = a[q];
          bool desc = ((i & ksz) == 0);
          bool sw = desc ? (A.s < B.s) : (A.s > B.s);
          if (sw) { a[i] = B; a[q] = A; }
        }
      }
      __syncthreads();
    }
  }
}

// ---------- greedy grouped NMS + top-1000 compaction ----------
__global__ __launch_bounds__(1024)
void nms_out(const SPair* __restrict__ npair,
             const float* __restrict__ cbox, const float* __restrict__ cscore,
             const int* __restrict__ cgroup, const int* __restrict__ cvalid,
             float* __restrict__ gbox, float* __restrict__ gscore,
             int* __restrict__ ggroup, int* __restrict__ gvalid,
             float* __restrict__ out) {
  const int im = blockIdx.x;
  const SPair* sp = npair + (size_t)im * NC;
  float* gb = gbox + (size_t)im * NC * 4;
  float* gs = gscore + (size_t)im * NC;
  int* gg = ggroup + (size_t)im * NC;
  int* gv = gvalid + (size_t)im * NC;

  // zero output slice
  for (int i = threadIdx.x; i < 5000; i += 1024) out[im * 5000 + i] = 0.f;
  // gather into score-sorted order
  for (int i = threadIdx.x; i < NC; i += 1024) {
    int slot = sp[i].id;
    int src = im * NC + slot;
    gb[i * 4 + 0] = cbox[src * 4 + 0];
    gb[i * 4 + 1] = cbox[src * 4 + 1];
    gb[i * 4 + 2] = cbox[src * 4 + 2];
    gb[i * 4 + 3] = cbox[src * 4 + 3];
    gs[i] = cscore[src];
    gg[i] = cgroup[src];
    gv[i] = cvalid[src];
  }
  __shared__ unsigned char supp[NC];
  for (int i = threadIdx.x; i < NC; i += 1024) supp[i] = 0;
  __shared__ int curKeep;
  __shared__ float bxs[4];
  __shared__ int bgrp;
  __syncthreads();

  const int NUSE = 3951;
  for (int i = 0; i < NUSE; ++i) {
    if (threadIdx.x == 0) {
      int k = (gv[i] && !supp[i]) ? 1 : 0;
      curKeep = k;
      gv[i] = k;  // re-purpose: final keep flag
      if (k) {
        bxs[0] = gb[i * 4 + 0]; bxs[1] = gb[i * 4 + 1];
        bxs[2] = gb[i * 4 + 2]; bxs[3] = gb[i * 4 + 3];
        bgrp = gg[i];
      }
    }
    __syncthreads();
    if (curKeep) {
      float x1 = bxs[0], y1 = bxs[1], x2 = bxs[2], y2 = bxs[3];
      float ai = (x2 - x1) * (y2 - y1);
      int g = bgrp;
      for (int j = i + 1 + threadIdx.x; j < NUSE; j += 1024) {
        if (gg[j] == g && !supp[j]) {
          float jx1 = gb[j * 4 + 0], jy1 = gb[j * 4 + 1];
          float jx2 = gb[j * 4 + 2], jy2 = gb[j * 4 + 3];
          float ix1 = fmaxf(x1, jx1), iy1 = fmaxf(y1, jy1);
          float ix2 = fminf(x2, jx2), iy2 = fminf(y2, jy2);
          float iw = fmaxf(ix2 - ix1, 0.f), ih = fmaxf(iy2 - iy1, 0.f);
          float inter = iw * ih;
          float aj = (jx2 - jx1) * (jy2 - jy1);
          float iou = inter / (ai + aj - inter + 1e-9f);
          if (iou > 0.7f) supp[j] = 1;
        }
      }
    }
    __syncthreads();
  }
  // compaction: first 1000 kept, in score order
  if (threadIdx.x == 0) {
    int c = 0;
    for (int i = 0; i < NUSE && c < 1000; ++i) {
      if (gv[i]) {
        float* o = out + (size_t)(im * 1000 + c) * 5;
        o[0] = gb[i * 4 + 0]; o[1] = gb[i * 4 + 1];
        o[2] = gb[i * 4 + 2]; o[3] = gb[i * 4 + 3];
        o[4] = gs[i];
        ++c;
      }
    }
  }
}

// ---------- host launch ----------
extern "C" void kernel_launch(void* const* d_in, const int* in_sizes, int n_in,
                              void* d_out, int out_size, void* d_ws, size_t ws_size,
                              hipStream_t stream) {
  (void)in_sizes; (void)n_in; (void)out_size; (void)ws_size;
  const float* fmaps[5]  = {(const float*)d_in[0], (const float*)d_in[2],
                            (const float*)d_in[4], (const float*)d_in[6],
                            (const float*)d_in[8]};
  const float* priors[5] = {(const float*)d_in[1], (const float*)d_in[3],
                            (const float*)d_in[5], (const float*)d_in[7],
                            (const float*)d_in[9]};
  const float* imsizes = (const float*)d_in[10];
  const float* conv_w  = (const float*)d_in[11];
  const float* conv_b  = (const float*)d_in[12];
  const float* log_w   = (const float*)d_in[13];
  const float* log_b   = (const float*)d_in[14];
  const float* reg_w   = (const float*)d_in[15];
  const float* reg_b   = (const float*)d_in[16];

  char* ws = (char*)d_ws;
  unsigned short* wpack  = (unsigned short*)(ws + OFF_WPACK);
  unsigned short* ppack  = (unsigned short*)(ws + OFF_PPACK);
  unsigned short* hbuf   = (unsigned short*)(ws + OFF_H);
  float*          projb  = (float*)(ws + OFF_PROJ);
  SPair*          sortb  = (SPair*)(ws + OFF_SORT);
  float*          cbox   = (float*)(ws + OFF_CBOX);
  float*          cscore = (float*)(ws + OFF_CSCORE);
  int*            cgroup = (int*)(ws + OFF_CGROUP);
  int*            cvalid = (int*)(ws + OFF_CVALID);
  SPair*          npair  = (SPair*)(ws + OFF_NPAIR);
  float*          gb     = (float*)(ws + OFF_GB);
  float*          gs     = (float*)(ws + OFF_GS);
  int*            gg     = (int*)(ws + OFF_GG);
  int*            gv     = (int*)(ws + OFF_GV);

  pack_conv_w<<<2304, 256, 0, stream>>>(conv_w, wpack);
  pack_proj_w<<<16, 256, 0, stream>>>(reg_w, log_w, ppack);

  const int Hs[5] = {100, 50, 25, 13, 7};
  const int Ws_[5] = {152, 76, 38, 19, 10};
  const int posOff[5] = {0, 30400, 38000, 39900, 40394};
  for (int l = 0; l < 5; ++l) {
    int pc = 2 * Hs[l] * Ws_[l];
    dim3 grid((pc + 63) / 64, 2);   // 128 oc per block
    conv3x3_bf16_wmma<<<grid, 256, 0, stream>>>(fmaps[l], wpack, conv_b, hbuf,
                                                Hs[l], Ws_[l], pc, posOff[l]);
  }

  proj_wmma<<<(NPOS_TOTAL + 127) / 128, 256, 0, stream>>>(hbuf, ppack, reg_b,
                                                          log_b, projb, NPOS_TOTAL);
  fill_sort<<<(2 * SORT_PER_IM) / 256, 256, 0, stream>>>(projb, sortb);
  sort_groups<<<10, 1024, 0, stream>>>(sortb);
  decode_combined<<<(2 * NC) / 256, 256, 0, stream>>>(
      sortb, projb, priors[0], priors[1], priors[2], priors[3], priors[4],
      imsizes, cbox, cscore, cgroup, cvalid, npair);
  sort_image<<<2, 1024, 0, stream>>>(npair);
  nms_out<<<2, 1024, 0, stream>>>(npair, cbox, cscore, cgroup, cvalid,
                                  gb, gs, gg, gv, (float*)d_out);
}
